// BatchedRouter_46548855554341
// MI455X (gfx1250) — matrix operand order repaired and checked by reading, not compiled, not run on hardware
//
#include <hip/hip_runtime.h>

typedef __attribute__((ext_vector_type(16))) _Float16 v16h;
typedef __attribute__((ext_vector_type(8)))  float    v8f;
typedef __attribute__((ext_vector_type(4)))  float    v4f;

#define NTOK   16384
#define DIM    4096
#define NEXP   64
#define TOPK   8
#define KSPLIT 4                 // waves cooperating on one 16-token group
#define KSLICE (DIM / KSPLIT)    // 1024 K per wave
#define GROUPS 2                 // 16-token groups per block
#define TOKPB  (GROUPS * 16)     // 32 tokens per block
#define LPAD   (NEXP + 4)        // LDS row padding to dodge bank conflicts

// 8 waves / block. Waves (g,q) for g in 0..1, q in 0..3: wave computes the
// partial 16-token x 64-expert logit tile of token-group g over K-quarter q.
// x is streamed exactly once from HBM (the bandwidth-critical operand, 268 MB
// -> ~11.7us floor at 23.3 TB/s); W (1 MB) is reread from L0/L2. The K-split
// raises the grid to 4096 waves so per-SIMD multi-wave occupancy hides the
// HBM latency of the x loads.
__global__ __launch_bounds__(256) void
router_wmma_kernel(const float* __restrict__ x,
                   const float* __restrict__ w,
                   float* __restrict__ probs_out,
                   int*   __restrict__ idx_out) {
  __shared__ float partial[GROUPS][KSPLIT][16][LPAD];

  const int tid  = threadIdx.x;
  const int wave = tid >> 5;
  const int lane = tid & 31;
  const int half = lane >> 4;   // 0: lanes 0-15, 1: lanes 16-31
  const int l16  = lane & 15;

  const int grp  = wave >> 2;   // token group within block
  const int kq   = wave & 3;    // K-quarter owned by this wave

  const int m_base = blockIdx.x * TOKPB + grp * 16;
  const float* xrow = x + (size_t)(m_base + l16) * DIM + kq * KSLICE;
  const float* wbase = w + kq * KSLICE;

  v8f acc[4] = {v8f{}, v8f{}, v8f{}, v8f{}};

  for (int k = 0; k < KSLICE; k += 32) {
    // ---- A fragment: 16x32 f16 tile, ISA layout:
    //   elem j (0..7)  -> K = k + 8*half + j
    //   elem j (8..15) -> K = k + 8*half + 16 + (j-8)
    const float* p0 = xrow + k + 8 * half;
    v4f x0 = *(const v4f*)(p0);
    v4f x1 = *(const v4f*)(p0 + 4);
    v4f x2 = *(const v4f*)(p0 + 16);
    v4f x3 = *(const v4f*)(p0 + 20);
    v16h a;
#pragma unroll
    for (int j = 0; j < 4; ++j) {
      a[j]      = (_Float16)x0[j];
      a[4 + j]  = (_Float16)x1[j];
      a[8 + j]  = (_Float16)x2[j];
      a[12 + j] = (_Float16)x3[j];
    }

    // ---- 4 B fragments (expert tiles) + WMMA accumulate.
    // B 32x16 f16 layout: col = lane%16, elem j (0..7) -> K = k + 8*half + j, elem j (8..15) -> K = k + 8*half + 16 + (j-8)
#pragma unroll
    for (int t = 0; t < 4; ++t) {
      const int e = t * 16 + l16;
      const float* wp = wbase + (size_t)e * DIM + k + 8 * half;
      v4f w0 = *(const v4f*)(wp);
      v4f w1 = *(const v4f*)(wp + 4);
      v4f w2 = *(const v4f*)(wp + 16);
      v4f w3 = *(const v4f*)(wp + 20);
      v16h b;
#pragma unroll
      for (int j = 0; j < 4; ++j) {
        b[j]      = (_Float16)w0[j];
        b[4 + j]  = (_Float16)w1[j];
        b[8 + j]  = (_Float16)w2[j];
        b[12 + j] = (_Float16)w3[j];
      }
      acc[t] = __builtin_amdgcn_wmma_f32_16x16x32_f16(
          /*neg_a=*/false, a, /*neg_b=*/false, b,
          /*c_mod=*/(short)0, acc[t], /*reuse_a=*/false, /*reuse_b=*/false);
    }
  }

  // ---- Dump partial logits to this wave's LDS slice.
  // C layout: VGPR v -> row (v + 8*half), col lane%16.
#pragma unroll
  for (int t = 0; t < 4; ++t) {
#pragma unroll
    for (int v = 0; v < 8; ++v) {
      partial[grp][kq][v + 8 * half][t * 16 + l16] = acc[t][v];
    }
  }

  __syncthreads();

  // ---- Per-token combine + softmax + top-8. Threads 0..31 own one token.
  if (tid < TOKPB) {
    const int g = tid >> 4;
    const int t = tid & 15;
    float* r0 = &partial[g][0][t][0];
    const float* r1 = &partial[g][1][t][0];
    const float* r2 = &partial[g][2][t][0];
    const float* r3 = &partial[g][3][t][0];

    // Deterministic K-quarter reduction (fixed summation order) + row max.
    float m = -3.402823466e+38f;
#pragma unroll 8
    for (int j = 0; j < NEXP; ++j) {
      float s = (r0[j] + r1[j]) + (r2[j] + r3[j]);
      r0[j] = s;
      m = fmaxf(m, s);
    }

    // exp(l - m); the 1/sum(exp) softmax factor cancels in top-k renorm and
    // the non-top-k outputs are zeroed, so it is never needed.
#pragma unroll 8
    for (int j = 0; j < NEXP; ++j) r0[j] = __expf(r0[j] - m);

    float vals[TOPK];
    int   idxs[TOPK];
    float ssum = 0.0f;
#pragma unroll
    for (int kk = 0; kk < TOPK; ++kk) {
      float bv = -1.0f;
      int   bi = 0;
#pragma unroll 8
      for (int j = 0; j < NEXP; ++j) {
        float pv = r0[j];
        if (pv > bv) { bv = pv; bi = j; }
      }
      vals[kk] = bv;
      idxs[kk] = bi;
      r0[bi] = -1.0f;  // exclude (exp values are > 0)
      ssum += bv;
    }
    const float inv = 1.0f / ssum;

    const int gtoken = blockIdx.x * TOKPB + tid;
    float* out = probs_out + (size_t)gtoken * NEXP;
    v4f z = {};
#pragma unroll
    for (int j = 0; j < NEXP / 4; ++j) *(v4f*)(out + 4 * j) = z;
#pragma unroll
    for (int kk = 0; kk < TOPK; ++kk) out[idxs[kk]] = vals[kk] * inv;
#pragma unroll
    for (int kk = 0; kk < TOPK; ++kk) idx_out[(size_t)gtoken * TOPK + kk] = idxs[kk];
  }
}

extern "C" void kernel_launch(void* const* d_in, const int* in_sizes, int n_in,
                              void* d_out, int out_size, void* d_ws, size_t ws_size,
                              hipStream_t stream) {
  (void)in_sizes; (void)n_in; (void)out_size; (void)d_ws; (void)ws_size;
  const float* x = (const float*)d_in[0];
  const float* w = (const float*)d_in[1];
  float* probs = (float*)d_out;
  int*   idx   = (int*)((float*)d_out + (size_t)NTOK * NEXP);  // second tuple output
  router_wmma_kernel<<<NTOK / TOKPB, 256, 0, stream>>>(x, w, probs, idx);
}
